// GroupedReadWrite_1005022347566
// MI455X (gfx1250) — matrix operand order repaired and checked by reading, not compiled, not run on hardware
//
#include <hip/hip_runtime.h>
#include <hip/hip_bf16.h>

// ---------------------------------------------------------------------------
// Grouped latent cross-attention for MI455X (gfx1250, wave32, WMMA).
//   1. f32->bf16 weight conversion (Wk, Wv, Wo)
//   2. mod = cond @ Wad.T + bad                        (tiny, VALU)
//   3. lat/q path: modulate, LN, Q-proj, head-LN       (32 rows, VALU)
//   4. LN(spatial) -> kvn bf16
//   5. fused K+V proj GEMM (WMMA bf16, async-to-LDS double-buffered staging)
//        K: +bias +per-head LN -> kh bf16 ; V: +bias -> vh bf16
//   6. group-restricted attention (16 keys/query)      -> attno bf16
//   7. O-proj GEMM (WMMA, async staging) + bo + lat    -> d_out f32
// ---------------------------------------------------------------------------

typedef __attribute__((ext_vector_type(16))) __bf16 v16bf;
typedef __attribute__((ext_vector_type(8)))  float  v8f;

union Frag16 { v16bf v; unsigned int u[8]; };

__device__ __forceinline__ float bf2f(unsigned short u) {
  unsigned int x = ((unsigned int)u) << 16;
  return __builtin_bit_cast(float, x);
}
__device__ __forceinline__ unsigned short f2bf(float f) {
  unsigned int x = __builtin_bit_cast(unsigned int, f);
  unsigned int r = (x + 0x7FFFu + ((x >> 16) & 1u)) >> 16;
  return (unsigned short)r;
}
__device__ __forceinline__ float wave_sum(float v) {
  for (int off = 16; off > 0; off >>= 1) v += __shfl_xor(v, off);
  return v;
}

// CDNA5 async global->LDS (ASYNCcnt-tracked, bypasses VGPRs).
// LDS operand: low 32 bits of the generic shared pointer == LDS byte address.
__device__ __forceinline__ void async_b128(unsigned lds, const void* g) {
  asm volatile("global_load_async_to_lds_b128 %0, %1, off"
               :: "v"(lds), "v"(g) : "memory");
}
#define WAIT_ASYNC(n) asm volatile("s_wait_asynccnt " #n ::: "memory")
#define LDSA(p) ((unsigned)(unsigned long long)(const void*)(p))

#define DMODEL 512
#define NTOK   4096
#define BATCH  4
#define GRPS   256
#define JLAT   8
#define KLAT   2048
#define MT     128
#define NT     64
#define KT     32

// ---------------------------------------------------------------------------
__global__ void f32_to_bf16_k(const float* __restrict__ x,
                              unsigned short* __restrict__ y, int n) {
  int i = blockIdx.x * 256 + threadIdx.x;
  if (i < n) y[i] = f2bf(x[i]);
}

__global__ __launch_bounds__(256)
void mod_k(const float* __restrict__ cond, const float* __restrict__ Wad,
           const float* __restrict__ bad, float* __restrict__ mod) {
  int idx = blockIdx.x * 256 + threadIdx.x;       // 0..4095
  int b = idx >> 10, o = idx & 1023;
  const float* cr = cond + b * DMODEL;
  const float* wr = Wad + (size_t)o * DMODEL;
  float acc = bad[o];
  for (int k = 0; k < DMODEL; ++k) acc += cr[k] * wr[k];
  mod[idx] = acc;
}

// Per (b,j): lat row (modulated, stored f32), LN, Q-proj, per-head LN -> qh bf16
__global__ __launch_bounds__(256)
void latq_k(const float* __restrict__ lt, const float* __restrict__ lp,
            const float* __restrict__ mod,
            const float* __restrict__ Wq, const float* __restrict__ bq,
            const float* __restrict__ nqg, const float* __restrict__ nqb,
            const float* __restrict__ qng, const float* __restrict__ qnb,
            float* __restrict__ lat, unsigned short* __restrict__ qh) {
  int j = blockIdx.x, b = blockIdx.y;
  int tid = threadIdx.x, wave = tid >> 5, lane = tid & 31;
  __shared__ float sLat[DMODEL], sQln[DMODEL], sQ[DMODEL];
  __shared__ float r1[8], r2[8];
  __shared__ float hMu[8], hRs[8];

  for (int i = tid; i < DMODEL; i += 256) {
    float v = (lt[j * DMODEL + i] + lp[j * DMODEL + i]) *
                  (1.0f + mod[b * 1024 + 512 + i]) + mod[b * 1024 + i];
    sLat[i] = v;
    lat[(size_t)(b * JLAT + j) * DMODEL + i] = v;
  }
  __syncthreads();
  float s = 0.f, s2 = 0.f;
  for (int i = tid; i < DMODEL; i += 256) { float v = sLat[i]; s += v; s2 += v * v; }
  s = wave_sum(s); s2 = wave_sum(s2);
  if (lane == 0) { r1[wave] = s; r2[wave] = s2; }
  __syncthreads();
  float tot = 0.f, tot2 = 0.f;
  for (int i = 0; i < 8; ++i) { tot += r1[i]; tot2 += r2[i]; }
  float mu = tot * (1.f / DMODEL);
  float rinv = rsqrtf(tot2 * (1.f / DMODEL) - mu * mu + 1e-5f);
  for (int i = tid; i < DMODEL; i += 256)
    sQln[i] = (sLat[i] - mu) * rinv * nqg[i] + nqb[i];
  __syncthreads();
  for (int half = 0; half < 2; ++half) {
    int c = tid + half * 256;
    const float* wr = Wq + (size_t)c * DMODEL;
    float acc = bq[c];
    for (int k = 0; k < DMODEL; ++k) acc += sQln[k] * wr[k];
    sQ[c] = acc;
  }
  __syncthreads();
  if (tid < 8) {
    float hs = 0.f, hs2 = 0.f;
    for (int d = 0; d < 64; ++d) { float v = sQ[tid * 64 + d]; hs += v; hs2 += v * v; }
    float hm = hs * (1.f / 64.f);
    hMu[tid] = hm;
    hRs[tid] = rsqrtf(hs2 * (1.f / 64.f) - hm * hm + 1e-5f);
  }
  __syncthreads();
  for (int half = 0; half < 2; ++half) {
    int c = tid + half * 256;
    int hd = c >> 6, d = c & 63;
    float v = (sQ[c] - hMu[hd]) * hRs[hd] * qng[d] + qnb[d];
    qh[(size_t)(b * JLAT + j) * DMODEL + c] = f2bf(v);
  }
}

// LayerNorm over rows of 512, f32 in -> bf16 out
__global__ __launch_bounds__(256)
void ln_rows_k(const float* __restrict__ x, const float* __restrict__ gg,
               const float* __restrict__ bb, unsigned short* __restrict__ y) {
  int row = blockIdx.x;
  int tid = threadIdx.x, wave = tid >> 5, lane = tid & 31;
  const float* xr = x + (size_t)row * DMODEL;
  __shared__ float r1[8], r2[8];
  float a0 = xr[tid], a1 = xr[tid + 256];
  float s = a0 + a1, s2 = a0 * a0 + a1 * a1;
  s = wave_sum(s); s2 = wave_sum(s2);
  if (lane == 0) { r1[wave] = s; r2[wave] = s2; }
  __syncthreads();
  float tot = 0.f, tot2 = 0.f;
  for (int i = 0; i < 8; ++i) { tot += r1[i]; tot2 += r2[i]; }
  float mu = tot * (1.f / DMODEL);
  float rinv = rsqrtf(tot2 * (1.f / DMODEL) - mu * mu + 1e-5f);
  unsigned short* yr = y + (size_t)row * DMODEL;
  yr[tid]       = f2bf((a0 - mu) * rinv * gg[tid] + bb[tid]);
  yr[tid + 256] = f2bf((a1 - mu) * rinv * gg[tid + 256] + bb[tid + 256]);
}

// ---------------------------------------------------------------------------
// Fragment helpers (16-bit WMMA operands, contiguous k order: element s of lane half h is k = 16h + s)
__device__ __forceinline__ void load_afrag(Frag16& af, const unsigned short* sA,
                                           int mrow, int khi) {
  for (int v = 0; v < 8; ++v)
    af.u[v] = *(const unsigned int*)&sA[mrow * KT + 2 * khi + 2 * v];
}
__device__ __forceinline__ void load_bfrag(Frag16& bf, const unsigned short* sB,
                                           int nloc, int klo) {
  for (int v = 0; v < 8; ++v)
    bf.u[v] = *(const unsigned int*)&sB[nloc * KT + klo + 2 * v];
}

// Fused K+V projection GEMM with per-head LN on K.
//   kh = headLN(kvn @ Wk^T + bk), vh = kvn @ Wv^T + bv
__global__ __launch_bounds__(256)
void gemm_kv_k(const unsigned short* __restrict__ A,
               const unsigned short* __restrict__ Wkb,
               const unsigned short* __restrict__ Wvb,
               const float* __restrict__ bk, const float* __restrict__ bv,
               const float* __restrict__ kng, const float* __restrict__ knb,
               unsigned short* __restrict__ kh, unsigned short* __restrict__ vh) {
  __shared__ unsigned short sA[2][MT * KT];   // 2 x 8 KB
  __shared__ unsigned short sBk[2][NT * KT];  // 2 x 4 KB
  __shared__ unsigned short sBv[2][NT * KT];  // 2 x 4 KB
  __shared__ float sC[MT * NT];               // 32 KB
  __shared__ float sMu[MT], sRs[MT];

  const int tid = threadIdx.x;
  const int wave = tid >> 5, lane = tid & 31;
  const int n0 = blockIdx.x * NT;
  const int m0 = blockIdx.y * MT;

  const int arow0 = (tid * 2) >> 2, akc0 = (tid * 2) & 3;      // uidx=2*tid,2*tid+1
  const int bn = tid >> 2, bkc = tid & 3;

  auto stage = [&](int buf, int k0) {
    // A tile: 128x32 bf16, 2 x b128 per thread (consecutive uidx)
    async_b128(LDSA(&sA[buf][arow0 * KT + akc0 * 8]),
               A + (size_t)(m0 + arow0) * DMODEL + k0 + akc0 * 8);
    {
      int uidx = tid * 2 + 1;
      int row = uidx >> 2, kc = uidx & 3;
      async_b128(LDSA(&sA[buf][row * KT + kc * 8]),
                 A + (size_t)(m0 + row) * DMODEL + k0 + kc * 8);
    }
    // B tiles: 64x32 bf16 each, 1 x b128 per thread
    async_b128(LDSA(&sBk[buf][bn * KT + bkc * 8]),
               Wkb + (size_t)(n0 + bn) * DMODEL + k0 + bkc * 8);
    async_b128(LDSA(&sBv[buf][bn * KT + bkc * 8]),
               Wvb + (size_t)(n0 + bn) * DMODEL + k0 + bkc * 8);
  };

  v8f ak[4], av[4];
  for (int i = 0; i < 4; ++i)
    for (int e = 0; e < 8; ++e) { ak[i][e] = 0.f; av[i][e] = 0.f; }

  const int mrow = wave * 16 + (lane & 15);
  const int khi = (lane < 16) ? 0 : 8;
  const int klo = (lane < 16) ? 0 : 16;

  stage(0, 0);
  for (int it = 0; it < DMODEL / KT; ++it) {
    int cur = it & 1;
    if (it + 1 < DMODEL / KT) {
      stage(cur ^ 1, (it + 1) * KT);
      WAIT_ASYNC(4);   // 4 newest (just issued) may remain; prior buffer done
    } else {
      WAIT_ASYNC(0);
    }
    __syncthreads();

    Frag16 af;
    load_afrag(af, sA[cur], mrow, khi);
    for (int ns = 0; ns < 4; ++ns) {
      Frag16 bfk, bfv;
      int nloc = ns * 16 + (lane & 15);
      load_bfrag(bfk, sBk[cur], nloc, klo);
      load_bfrag(bfv, sBv[cur], nloc, klo);
      ak[ns] = __builtin_amdgcn_wmma_f32_16x16x32_bf16(
          false, af.v, false, bfk.v, (short)0, ak[ns], false, false);
      av[ns] = __builtin_amdgcn_wmma_f32_16x16x32_bf16(
          false, af.v, false, bfv.v, (short)0, av[ns], false, false);
    }
    __syncthreads();   // protect buffer about to be overwritten next iteration
  }

  // ---- K epilogue: spill, per-head LN (bias folded into stats), store ----
  for (int ns = 0; ns < 4; ++ns) {
    int ncol = ns * 16 + (lane & 15);
    for (int r = 0; r < 8; ++r) {
      int mr = wave * 16 + ((lane < 16) ? r : 8 + r);
      sC[mr * NT + ncol] = ak[ns][r];
    }
  }
  __syncthreads();
  if (tid < MT) {
    float s = 0.f, s2 = 0.f;
    for (int c = 0; c < NT; ++c) {
      float v = sC[tid * NT + c] + bk[n0 + c];
      s += v; s2 += v * v;
    }
    float mu = s * (1.f / NT);
    sMu[tid] = mu;
    sRs[tid] = rsqrtf(s2 * (1.f / NT) - mu * mu + 1e-5f);
  }
  __syncthreads();
  for (int i = 0; i < 32; ++i) {
    int idx = i * 256 + tid;
    int row = idx >> 6, col = idx & 63;
    float v = (sC[idx] + bk[n0 + col] - sMu[row]) * sRs[row] * kng[col] + knb[col];
    kh[(size_t)(m0 + row) * DMODEL + n0 + col] = f2bf(v);
  }
  __syncthreads();

  // ---- V epilogue: spill, +bias, store ----
  for (int ns = 0; ns < 4; ++ns) {
    int ncol = ns * 16 + (lane & 15);
    for (int r = 0; r < 8; ++r) {
      int mr = wave * 16 + ((lane < 16) ? r : 8 + r);
      sC[mr * NT + ncol] = av[ns][r];
    }
  }
  __syncthreads();
  for (int i = 0; i < 32; ++i) {
    int idx = i * 256 + tid;
    int row = idx >> 6, col = idx & 63;
    float v = sC[idx] + bv[n0 + col];
    vh[(size_t)(m0 + row) * DMODEL + n0 + col] = f2bf(v);
  }
}

// O-projection GEMM: d_out = attno @ Wo^T + bo + lat(repeated over groups)
__global__ __launch_bounds__(256)
void gemm_o_k(const unsigned short* __restrict__ A,
              const unsigned short* __restrict__ Wob,
              const float* __restrict__ bo, const float* __restrict__ lat,
              float* __restrict__ outf) {
  __shared__ unsigned short sA[2][MT * KT];
  __shared__ unsigned short sB[2][NT * KT];
  __shared__ float sC[MT * NT];

  const int tid = threadIdx.x;
  const int wave = tid >> 5, lane = tid & 31;
  const int n0 = blockIdx.x * NT;
  const int m0 = blockIdx.y * MT;

  const int arow0 = (tid * 2) >> 2, akc0 = (tid * 2) & 3;
  const int bn = tid >> 2, bkc = tid & 3;

  auto stage = [&](int buf, int k0) {
    async_b128(LDSA(&sA[buf][arow0 * KT + akc0 * 8]),
               A + (size_t)(m0 + arow0) * DMODEL + k0 + akc0 * 8);
    {
      int uidx = tid * 2 + 1;
      int row = uidx >> 2, kc = uidx & 3;
      async_b128(LDSA(&sA[buf][row * KT + kc * 8]),
                 A + (size_t)(m0 + row) * DMODEL + k0 + kc * 8);
    }
    async_b128(LDSA(&sB[buf][bn * KT + bkc * 8]),
               Wob + (size_t)(n0 + bn) * DMODEL + k0 + bkc * 8);
  };

  v8f acc[4];
  for (int i = 0; i < 4; ++i)
    for (int e = 0; e < 8; ++e) acc[i][e] = 0.f;

  const int mrow = wave * 16 + (lane & 15);
  const int khi = (lane < 16) ? 0 : 8;
  const int klo = (lane < 16) ? 0 : 16;

  stage(0, 0);
  for (int it = 0; it < DMODEL / KT; ++it) {
    int cur = it & 1;
    if (it + 1 < DMODEL / KT) {
      stage(cur ^ 1, (it + 1) * KT);
      WAIT_ASYNC(3);
    } else {
      WAIT_ASYNC(0);
    }
    __syncthreads();

    Frag16 af;
    load_afrag(af, sA[cur], mrow, khi);
    for (int ns = 0; ns < 4; ++ns) {
      Frag16 bf;
      int nloc = ns * 16 + (lane & 15);
      load_bfrag(bf, sB[cur], nloc, klo);
      acc[ns] = __builtin_amdgcn_wmma_f32_16x16x32_bf16(
          false, af.v, false, bf.v, (short)0, acc[ns], false, false);
    }
    __syncthreads();
  }

  for (int ns = 0; ns < 4; ++ns) {
    int ncol = ns * 16 + (lane & 15);
    for (int r = 0; r < 8; ++r) {
      int mr = wave * 16 + ((lane < 16) ? r : 8 + r);
      sC[mr * NT + ncol] = acc[ns][r];
    }
  }
  __syncthreads();
  for (int i = 0; i < 32; ++i) {
    int idx = i * 256 + tid;
    int row = idx >> 6, col = idx & 63;
    int mg = m0 + row;
    int b = mg >> 11;   // / KLAT
    int j = mg & 7;     // latent index within group
    float v = sC[idx] + bo[n0 + col] +
              lat[(size_t)((b << 3) | j) * DMODEL + n0 + col];
    outf[(size_t)mg * DMODEL + n0 + col] = v;
  }
}

// ---------------------------------------------------------------------------
// Group-restricted attention: block = (g, b), wave = head. 16 keys per query.
__global__ __launch_bounds__(256)
void attn_k(const unsigned short* __restrict__ qh,
            const unsigned short* __restrict__ kh,
            const unsigned short* __restrict__ vh,
            unsigned short* __restrict__ attno) {
  int gid = blockIdx.x, b = blockIdx.y;
  int h = threadIdx.x >> 5, lane = threadIdx.x & 31;
  int gr = gid >> 4, gc = gid & 15;
  __shared__ float sP[8][8][16];

  int qi = lane >> 2;
  int kq = lane & 3;
  const unsigned short* qrow = qh + ((size_t)(b * JLAT + qi) * DMODEL + h * 64);

  float sc[4];
  for (int t = 0; t < 4; ++t) {
    int ki = kq * 4 + t;
    int n = (gr * 4 + (ki >> 2)) * 64 + gc * 4 + (ki & 3);
    const unsigned short* krow = kh + ((size_t)(b * NTOK + n) * DMODEL + h * 64);
    float acc = 0.f;
    for (int d = 0; d < 64; ++d) acc += bf2f(qrow[d]) * bf2f(krow[d]);
    sc[t] = acc * 0.125f;  // 1/sqrt(64)
  }
  float mx = fmaxf(fmaxf(sc[0], sc[1]), fmaxf(sc[2], sc[3]));
  mx = fmaxf(mx, __shfl_xor(mx, 1));
  mx = fmaxf(mx, __shfl_xor(mx, 2));
  float se = 0.f;
  for (int t = 0; t < 4; ++t) { sc[t] = __expf(sc[t] - mx); se += sc[t]; }
  se += __shfl_xor(se, 1);
  se += __shfl_xor(se, 2);
  float inv = 1.0f / se;
  for (int t = 0; t < 4; ++t) sP[h][qi][kq * 4 + t] = sc[t] * inv;
  __syncthreads();

  int dseg = lane & 3;
  float o[16];
  for (int d = 0; d < 16; ++d) o[d] = 0.f;
  for (int ki = 0; ki < 16; ++ki) {
    int n = (gr * 4 + (ki >> 2)) * 64 + gc * 4 + (ki & 3);
    const unsigned short* vrow =
        vh + ((size_t)(b * NTOK + n) * DMODEL + h * 64 + dseg * 16);
    float p = sP[h][qi][ki];
    for (int d = 0; d < 16; ++d) o[d] += p * bf2f(vrow[d]);
  }
  unsigned short* orow =
      attno + ((size_t)(b * KLAT + gid * JLAT + qi) * DMODEL + h * 64 + dseg * 16);
  for (int d = 0; d < 16; ++d) orow[d] = f2bf(o[d]);
}

// ---------------------------------------------------------------------------
extern "C" void kernel_launch(void* const* d_in, const int* in_sizes, int n_in,
                              void* d_out, int out_size, void* d_ws, size_t ws_size,
                              hipStream_t stream) {
  const float* spatial = (const float*)d_in[0];
  const float* cond    = (const float*)d_in[1];
  const float* lt      = (const float*)d_in[2];
  const float* lp      = (const float*)d_in[3];
  const float* Wq      = (const float*)d_in[4];
  const float* bq      = (const float*)d_in[5];
  const float* Wk      = (const float*)d_in[6];
  const float* bk      = (const float*)d_in[7];
  const float* Wv      = (const float*)d_in[8];
  const float* bv      = (const float*)d_in[9];
  const float* Wo      = (const float*)d_in[10];
  const float* bo      = (const float*)d_in[11];
  const float* Wad     = (const float*)d_in[12];
  const float* bad     = (const float*)d_in[13];
  const float* qn_g    = (const float*)d_in[14];
  const float* qn_b    = (const float*)d_in[15];
  const float* kn_g    = (const float*)d_in[16];
  const float* kn_b    = (const float*)d_in[17];
  const float* nq_g    = (const float*)d_in[18];
  const float* nq_b    = (const float*)d_in[19];
  const float* nkv_g   = (const float*)d_in[20];
  const float* nkv_b   = (const float*)d_in[21];
  (void)in_sizes; (void)n_in; (void)out_size; (void)ws_size;

  char* ws = (char*)d_ws;
  size_t off = 0;
  auto carve = [&](size_t bytes) {
    void* p = ws + off;
    off = (off + bytes + 255) & ~(size_t)255;
    return p;
  };
  float*          mod  = (float*)carve(4096 * 4);
  float*          lat  = (float*)carve((size_t)BATCH * JLAT * DMODEL * 4);
  unsigned short* qh   = (unsigned short*)carve((size_t)BATCH * JLAT * DMODEL * 2);
  unsigned short* Wkb  = (unsigned short*)carve((size_t)DMODEL * DMODEL * 2);
  unsigned short* Wvb  = (unsigned short*)carve((size_t)DMODEL * DMODEL * 2);
  unsigned short* Wob  = (unsigned short*)carve((size_t)DMODEL * DMODEL * 2);
  unsigned short* kvn  = (unsigned short*)carve((size_t)BATCH * NTOK * DMODEL * 2);
  unsigned short* kh   = (unsigned short*)carve((size_t)BATCH * NTOK * DMODEL * 2);
  unsigned short* vh   = (unsigned short*)carve((size_t)BATCH * NTOK * DMODEL * 2);
  unsigned short* attno = kvn;  // alias: kvn dead after fused K/V GEMM

  const int WN = DMODEL * DMODEL;
  f32_to_bf16_k<<<(WN + 255) / 256, 256, 0, stream>>>(Wk, Wkb, WN);
  f32_to_bf16_k<<<(WN + 255) / 256, 256, 0, stream>>>(Wv, Wvb, WN);
  f32_to_bf16_k<<<(WN + 255) / 256, 256, 0, stream>>>(Wo, Wob, WN);

  mod_k<<<16, 256, 0, stream>>>(cond, Wad, bad, mod);
  latq_k<<<dim3(JLAT, BATCH), 256, 0, stream>>>(lt, lp, mod, Wq, bq,
                                                nq_g, nq_b, qn_g, qn_b, lat, qh);
  ln_rows_k<<<BATCH * NTOK, 256, 0, stream>>>(spatial, nkv_g, nkv_b, kvn);

  // Fused K+V projection (+ per-head LN on K)
  gemm_kv_k<<<dim3(8, BATCH * NTOK / MT), 256, 0, stream>>>(
      kvn, Wkb, Wvb, bk, bv, kn_g, kn_b, kh, vh);

  attn_k<<<dim3(GRPS, BATCH), 256, 0, stream>>>(qh, kh, vh, attno);

  // O projection + lat + bo -> f32 output
  gemm_o_k<<<dim3(8, BATCH * KLAT / MT), 256, 0, stream>>>(
      attno, Wob, bo, lat, (float*)d_out);
}
